// SDPAModule_446676599323
// MI455X (gfx1250) — matrix operand order repaired and checked by reading, not compiled, not run on hardware
//
#include <hip/hip_runtime.h>

typedef __attribute__((ext_vector_type(16))) _Float16 v16h;
typedef __attribute__((ext_vector_type(8)))  float    v8f;

#define BATCH   2
#define SEQ     2048
#define NHEADS  32
#define NKV     8
#define HDIM    64
#define WINDOW  128
#define QTILE   16
#define NQT     (SEQ / QTILE)   // 128 query tiles
#define NKT     9               // key tiles covering the 128-window for a 16-q tile
#define PCOLS   160             // padded key columns: 5 chunks * 32
#define WAVES   8               // waves per block (256 threads) = 8 heads per block

__device__ __forceinline__ float red_max16(float v) {
#pragma unroll
  for (int off = 1; off < 16; off <<= 1)
    v = fmaxf(v, __shfl_xor(v, off, 32));
  return v;
}
__device__ __forceinline__ float red_sum16(float v) {
#pragma unroll
  for (int off = 1; off < 16; off <<= 1)
    v += __shfl_xor(v, off, 32);
  return v;
}

__global__ __launch_bounds__(WAVES * 32) void swa_attn_kernel(
    const float* __restrict__ Q, const float* __restrict__ K,
    const float* __restrict__ V, float* __restrict__ O)
{
  __shared__ _Float16 ldsP[WAVES * QTILE * PCOLS];  // 8*16*160*2 = 40960 B

  const int tid  = threadIdx.x;
  const int wave = tid >> 5;
  const int lane = tid & 31;
  const int m    = lane & 15;   // row (A) or column (B/C) index within tile
  const int hi   = lane >> 4;   // which 16-lane half
  const int dmh  = m - hi * 8;  // boundary-mask key coordinate, in [-8, 15]

  // Scalar (blockIdx-derived) task coordinates: qt and b are wave-uniform AND
  // visible as uniform to the compiler -> scalar branches around the WMMAs.
  const int bx = blockIdx.x;
  const int qt = bx % NQT;                 // query tile (scalar)
  const int hg = (bx / NQT) % (NHEADS / WAVES);
  const int b  = bx / (NQT * (NHEADS / WAVES));
  const int h  = hg * WAVES + wave;        // head: one per wave
  const int kv = h >> 2;                   // h / Q_MULT
  const int q0 = qt * QTILE;

  // sm_scale * log2(e), folded into Q before f16 conversion: scores come out
  // of the QK WMMA already in the log2 domain.
  const float sm_l2 = 0.125f * 1.4426950408889634f;

  // ---- Q tile in A layout: lane m = row, element e -> dim (e&7)+((e&8)<<1)+8*hi ----
  const float* qrow = Q + ((size_t)b * SEQ + q0 + m) * (NHEADS * HDIM) + h * HDIM;
  v16h Aq0, Aq1;
#pragma unroll
  for (int e = 0; e < 16; ++e) {
    const int d = (e & 7) + ((e & 8) << 1) + hi * 8;
    Aq0[e] = (_Float16)(qrow[d] * sm_l2);
    Aq1[e] = (_Float16)(qrow[32 + d] * sm_l2);
  }

  // ---- scores: 9 key tiles of 16, kept in registers (C layout: row = r + 8*hi, col = m)
  // Mask geometry: only kt==0 (window lower edge) and kt==8 (causal diagonal)
  // ever need masking, and both reduce to comparing dmh with r (complementary).
  float s[NKT][8];
  const int kt0 = qt - (NKT - 1);          // scalar
#pragma unroll
  for (int kt = 0; kt < NKT; ++kt) {
    const int ktg = kt0 + kt;
    if (ktg < 0) {                         // scalar branch: EXEC untouched
#pragma unroll
      for (int r = 0; r < 8; ++r) s[kt][r] = -__builtin_inff();
      continue;
    }
    const int k0 = ktg * QTILE;
    // K tile as B operand: lane m = key column, element e -> dim (e&7)+((e&8)<<1)+8*hi
    const float* krow = K + ((size_t)b * SEQ + k0 + m) * (NKV * HDIM) + kv * HDIM;
    v16h B0, B1;
#pragma unroll
    for (int e = 0; e < 16; ++e) {
      const int d = (e & 7) + ((e & 8) << 1) + hi * 8;
      B0[e] = (_Float16)krow[d];
      B1[e] = (_Float16)krow[32 + d];
    }
    v8f c = {};
    c = __builtin_amdgcn_wmma_f32_16x16x32_f16(false, Aq0, false, B0, (short)0, c, false, false);
    c = __builtin_amdgcn_wmma_f32_16x16x32_f16(false, Aq1, false, B1, (short)0, c, false, false);
#pragma unroll
    for (int r = 0; r < 8; ++r) {
      if (kt == 0)      s[kt][r] = (dmh >  r) ? c[r] : -__builtin_inff(); // k > q-128
      else if (kt == 8) s[kt][r] = (dmh <= r) ? c[r] : -__builtin_inff(); // k <= q
      else              s[kt][r] = c[r];                                  // fully inside
    }
  }

  // ---- exact softmax over the full 144-key window (logits already log2-scaled) ----
  float lrow[8];
#pragma unroll
  for (int r = 0; r < 8; ++r) {
    float mv = -__builtin_inff();
#pragma unroll
    for (int kt = 0; kt < NKT; ++kt) mv = fmaxf(mv, s[kt][r]);
    mv = red_max16(mv);          // row always contains k==q, so mv is finite
    float ls = 0.0f;
#pragma unroll
    for (int kt = 0; kt < NKT; ++kt) {
      const float p = __builtin_amdgcn_exp2f(s[kt][r] - mv);
      s[kt][r] = p;
      ls += p;
    }
    lrow[r] = red_sum16(ls);
  }

  // ---- stage P (f16) through LDS to transpose C layout -> A layout ----
  _Float16* myP = ldsP + wave * (QTILE * PCOLS);
#pragma unroll
  for (int kt = 0; kt < NKT; ++kt) {
    const int col = kt * 16 + m;
#pragma unroll
    for (int r = 0; r < 8; ++r)
      myP[(r + hi * 8) * PCOLS + col] = (_Float16)s[kt][r];
  }
#pragma unroll
  for (int r = 0; r < 8; ++r)   // zero the pad columns 144..159
    myP[(r + hi * 8) * PCOLS + NKT * 16 + m] = (_Float16)0.0f;
  asm volatile("s_wait_dscnt 0" ::: "memory");

  // ---- P*V: 5 chunks of K=32 keys x 4 dim tiles of N=16 ----
  v8f o0 = {}, o1 = {}, o2 = {}, o3 = {};
  const int kfirst = kt0 * QTILE;  // q0 - 128 (scalar)
#pragma unroll
  for (int ci = 0; ci < 5; ++ci) {
    // A operand: P chunk, lane m = query row, element e -> key (e&7)+((e&8)<<1)+8*hi
    const _Float16* prow = myP + m * PCOLS + ci * 32;
    v16h Ap;
#pragma unroll
    for (int e = 0; e < 16; ++e)
      Ap[e] = prow[(e & 7) + ((e & 8) << 1) + hi * 8];

    const int kb = kfirst + ci * 32;
#pragma unroll
    for (int nt = 0; nt < 4; ++nt) {
      // B operand: V chunk, lane m -> dim column nt*16+m, element e -> key (e&7)+((e&8)<<1)+8*hi
      const int n = nt * 16 + m;
      v16h Bv;
#pragma unroll
      for (int e = 0; e < 16; ++e) {
        int kg = kb + (e & 7) + ((e & 8) << 1) + hi * 8;
        kg = kg < 0 ? 0 : (kg > SEQ - 1 ? SEQ - 1 : kg);  // clamped; P==0 on those cols
        Bv[e] = (_Float16)V[((size_t)b * SEQ + kg) * (NKV * HDIM) + kv * HDIM + n];
      }
      if (nt == 0) o0 = __builtin_amdgcn_wmma_f32_16x16x32_f16(false, Ap, false, Bv, (short)0, o0, false, false);
      if (nt == 1) o1 = __builtin_amdgcn_wmma_f32_16x16x32_f16(false, Ap, false, Bv, (short)0, o1, false, false);
      if (nt == 2) o2 = __builtin_amdgcn_wmma_f32_16x16x32_f16(false, Ap, false, Bv, (short)0, o2, false, false);
      if (nt == 3) o3 = __builtin_amdgcn_wmma_f32_16x16x32_f16(false, Ap, false, Bv, (short)0, o3, false, false);
    }
  }

  // ---- normalize and store (C layout: row = r + 8*hi, dim col = nt*16 + m) ----
  float inv[8];
#pragma unroll
  for (int r = 0; r < 8; ++r) inv[r] = 1.0f / lrow[r];

#pragma unroll
  for (int r = 0; r < 8; ++r) {
    const size_t row = ((size_t)b * SEQ + (q0 + r + hi * 8)) * (NHEADS * HDIM) + h * HDIM;
    O[row + 0  + m] = o0[r] * inv[r];
    O[row + 16 + m] = o1[r] * inv[r];
    O[row + 32 + m] = o2[r] * inv[r];
    O[row + 48 + m] = o3[r] * inv[r];
  }
}

extern "C" void kernel_launch(void* const* d_in, const int* in_sizes, int n_in,
                              void* d_out, int out_size, void* d_ws, size_t ws_size,
                              hipStream_t stream) {
  const float* Q = (const float*)d_in[0];
  const float* K = (const float*)d_in[1];
  const float* V = (const float*)d_in[2];
  // d_in[3] = sinks: dead code in the reference math, intentionally unused.
  float* O = (float*)d_out;

  // blocks enumerate (batch, head-octet, query-tile) so qt/b are scalar;
  // the 8 waves in a block cover 8 consecutive heads at the same qt.
  const int blocks = BATCH * (NHEADS / WAVES) * NQT;  // 2*4*128 = 1024
  swa_attn_kernel<<<blocks, WAVES * 32, 0, stream>>>(Q, K, V, O);
}
